// T5Gemma2MergedAttention_84069689852128
// MI455X (gfx1250) — compile-verified, matrix operand order checked by reading, not run on hardware
//
#include <hip/hip_runtime.h>

#define BB   2
#define TT   1024
#define SENC 1024
#define STOT 2048
#define DD   2048
#define NQh  16
#define NKVh 8
#define HH   256
#define CAPF 50.0f
#define SCALEF 0.0625f
#define KMASK -2.3819763e+38f

typedef __attribute__((ext_vector_type(16))) __bf16 v16bf;
typedef __attribute__((ext_vector_type(8)))  float  v8f;
typedef __attribute__((ext_vector_type(4)))  unsigned int u32x4;

union FragU { u32x4 u[2]; v16bf v; };

__device__ __forceinline__ v8f wmma_bf16(v16bf a, v16bf b, v8f c) {
  return __builtin_amdgcn_wmma_f32_16x16x32_bf16(false, a, false, b, (short)0, c,
                                                 false, false);
}

// Load a 16x32 A-style (or B-from-transposed) bf16 fragment.
// base points at (row 0, k 0); rows have stride ld (elements), k contiguous.
__device__ __forceinline__ v16bf load_frag(const __bf16* base, int ld, int lane) {
  const int m = lane & 15;
  const int khalf = (lane >> 4) * 8;
  const __bf16* p = base + (size_t)m * ld;
  FragU f;
  f.u[0] = *reinterpret_cast<const u32x4*>(p + khalf);
  f.u[1] = *reinterpret_cast<const u32x4*>(p + 16 + khalf);
  return f.v;
}

// ---------------- elementwise cast f32 -> bf16 ----------------
__global__ void cast_bf16_kernel(const float* __restrict__ src,
                                 __bf16* __restrict__ dst, int n) {
  int i = blockIdx.x * 256 + threadIdx.x;
  if (i < n) dst[i] = (__bf16)src[i];
}

// -------- transpose+cast: src f32 [R,C] (per z-slice) -> dst bf16 [C,R] -----
__global__ void transpose_cast_kernel(const float* __restrict__ src,
                                      __bf16* __restrict__ dst, int R, int C) {
  size_t slice = (size_t)R * C;
  const float* s = src + blockIdx.z * slice;
  __bf16* d = dst + blockIdx.z * slice;
  int i = blockIdx.x * 256 + threadIdx.x;
  if (i < R * C) {
    int r = i / C, c = i % C;
    d[(size_t)c * R + r] = (__bf16)s[i];
  }
}

// ------- generic bf16 WMMA GEMM: C[M,N] = A[M,K] * Bt[N,K]^T ----------------
// One wave computes a 64x32 f32 tile (4x2 register-blocked 16x16 WMMA tiles):
// per K-step (32): 4 A-frags + 2 B-frags (12x b128 loads) feed 8 WMMAs.
// launch_bounds(32,1): single wave per block, allow full VGPR budget (no spill).
__global__ void __launch_bounds__(32, 1)
gemm_bf16_wmma(const __bf16* __restrict__ A, int lda, size_t az,
               const __bf16* __restrict__ Bt, int ldbt, size_t bz,
               float* __restrict__ C, int ldc, size_t cz, int K) {
  const int lane = threadIdx.x;
  const int m16 = lane & 15;
  const int khalf = (lane >> 4) * 8;
  const __bf16* abase = A + blockIdx.z * az + ((size_t)blockIdx.x * 64 + m16) * lda;
  const __bf16* bbase = Bt + blockIdx.z * bz + ((size_t)blockIdx.y * 32 + m16) * ldbt;
  v8f acc[4][2] = {};
  for (int k0 = 0; k0 < K; k0 += 32) {
    v16bf af[4], bf[2];
#pragma unroll
    for (int i = 0; i < 4; ++i) {
      FragU fa;
      const __bf16* ap = abase + (size_t)(i * 16) * lda + k0;
      fa.u[0] = *reinterpret_cast<const u32x4*>(ap + khalf);
      fa.u[1] = *reinterpret_cast<const u32x4*>(ap + 16 + khalf);
      af[i] = fa.v;
    }
#pragma unroll
    for (int j = 0; j < 2; ++j) {
      FragU fb;
      const __bf16* bp = bbase + (size_t)(j * 16) * ldbt + k0;
      fb.u[0] = *reinterpret_cast<const u32x4*>(bp + khalf);
      fb.u[1] = *reinterpret_cast<const u32x4*>(bp + 16 + khalf);
      bf[j] = fb.v;
    }
#pragma unroll
    for (int i = 0; i < 4; ++i)
#pragma unroll
      for (int j = 0; j < 2; ++j)
        acc[i][j] = wmma_bf16(af[i], bf[j], acc[i][j]);
  }
  const int mrow = (lane >> 4) * 8;
  float* cbase =
      C + blockIdx.z * cz + (size_t)blockIdx.x * 64 * ldc + blockIdx.y * 32;
#pragma unroll
  for (int i = 0; i < 4; ++i)
#pragma unroll
    for (int j = 0; j < 2; ++j)
#pragma unroll
      for (int r = 0; r < 8; ++r)
        cbase[(size_t)(i * 16 + mrow + r) * ldc + j * 16 + m16] = acc[i][j][r];
}

// -------- RMS-norm (+optional RoPE) epilogue, writes bf16 [b, n, s_off+t, h] --
// grid = (Tloc, Nh, B), block = 256 (= H)
__global__ void qk_epilogue(const float* __restrict__ x,   // [B,Tloc,Nh,H]
                            const float* __restrict__ sc,  // [H]
                            const int* __restrict__ pos,   // [B,T]
                            float mul, int use_rope,
                            __bf16* __restrict__ out, int s_len, int s_off) {
  const int t = blockIdx.x, n = blockIdx.y, b = blockIdx.z, h = threadIdx.x;
  const int Tloc = gridDim.x, Nh = gridDim.y;
  const float* row = x + (((size_t)b * Tloc + t) * Nh + n) * HH;
  const float v = row[h];
  __shared__ float red[HH];
  red[h] = v * v;
  __syncthreads();
  for (int s = 128; s > 0; s >>= 1) {
    if (h < s) red[h] += red[h + s];
    __syncthreads();
  }
  const float rs = rsqrtf(red[0] * (1.0f / 256.0f) + 1e-6f);
  float res = v * rs * (1.0f + sc[h]);
  if (use_rope) {
    const int hp = h & 127;
    const int hq = (h < 128) ? h + 128 : h - 128;
    const float xo = row[hq] * rs * (1.0f + sc[hq]);
    const float ts = __powf(10000.0f, (float)hp * (1.0f / 128.0f));
    const float ph = (float)pos[(size_t)b * Tloc + t] / ts;
    const float sn = __sinf(ph), cs = __cosf(ph);
    res = (h < 128) ? (res * cs - xo * sn) : (res * cs + xo * sn);
  }
  out[(((size_t)b * Nh + n) * s_len + (s_off + t)) * HH + h] = (__bf16)(res * mul);
}

// -------- pack V (self+cross f32) into transposed bf16 vt[b][kv][h][s] ------
__global__ void v_pack_kernel(const float* __restrict__ selfv,   // [B,T,NKV,H]
                              const float* __restrict__ crossv,  // [B,SENC,NKV,H]
                              __bf16* __restrict__ vt) {
  size_t i = (size_t)blockIdx.x * 256 + threadIdx.x;
  if (i >= (size_t)BB * STOT * NKVh * HH) return;
  int h = i & 255;
  int kv = (i >> 8) & 7;
  int s = (i >> 11) & 2047;
  int b = (int)(i >> 22);
  float val = (s < TT)
      ? selfv[(((size_t)b * TT + s) * NKVh + kv) * HH + h]
      : crossv[(((size_t)b * SENC + (s - TT)) * NKVh + kv) * HH + h];
  vt[(((size_t)b * NKVh + kv) * HH + h) * (size_t)STOT + s] = (__bf16)val;
}

// ---------------- flash attention, one wave = 16 queries x 128 h-half -------
// grid = (T/16, NQ, B*2), block = 32
__global__ void __launch_bounds__(32, 1)
attn_wmma(const __bf16* __restrict__ qa,  // [B,NQ,T,H]
          const __bf16* __restrict__ ka,  // [B,NKV,STOT,H]
          const __bf16* __restrict__ vt,  // [B,NKV,H,STOT]
          __bf16* __restrict__ oa) {      // [B*T, NQ*H]
  const int lane = threadIdx.x;
  const int qt = blockIdx.x;
  const int n = blockIdx.y;
  const int b = blockIdx.z >> 1;
  const int hh = blockIdx.z & 1;
  const int kv = n >> 1;  // G = NQ/NKV = 2
  const int t0 = qt * 16;
  const int m16 = lane & 15;
  const int khalf = (lane >> 4) * 8;
  const int mrow = (lane >> 4) * 8;

  const __bf16* Q = qa + (((size_t)b * NQh + n) * TT + t0) * HH;
  const __bf16* Kp = ka + ((size_t)b * NKVh + kv) * (size_t)STOT * HH;
  const __bf16* Vp = vt + (((size_t)b * NKVh + kv) * HH + hh * 128) * (size_t)STOT;

  v16bf qf[8];
#pragma unroll
  for (int j = 0; j < 8; ++j) qf[j] = load_frag(Q + j * 32, HH, lane);

  v8f O[8] = {};
  float mrun[8], lrun[8];
#pragma unroll
  for (int r = 0; r < 8; ++r) {
    mrun[r] = -__builtin_inff();
    lrun[r] = 0.0f;
  }

  __shared__ __attribute__((aligned(16))) __bf16 P[16 * 32];

  for (int s0 = 0; s0 < STOT; s0 += 32) {
    v8f L0 = {}, L1 = {};
#pragma unroll
    for (int j = 0; j < 8; ++j) {
      v16bf k0f = load_frag(Kp + (size_t)s0 * HH + j * 32, HH, lane);
      v16bf k1f = load_frag(Kp + (size_t)(s0 + 16) * HH + j * 32, HH, lane);
      L0 = wmma_bf16(qf[j], k0f, L0);
      L1 = wmma_bf16(qf[j], k1f, L1);
    }
    float alpha[8];
#pragma unroll
    for (int r = 0; r < 8; ++r) {
      const int tg = t0 + mrow + r;
      const int sk0 = s0 + m16;
      float x0 = CAPF * tanhf(L0[r] * (1.0f / CAPF));
      float x1 = CAPF * tanhf(L1[r] * (1.0f / CAPF));
      if (sk0 < TT && sk0 > tg) x0 = KMASK;            // causal (self half)
      if (sk0 + 16 < TT && sk0 + 16 > tg) x1 = KMASK;  // cross half always visible
      float mx = fmaxf(x0, x1);
#pragma unroll
      for (int d = 1; d < 16; d <<= 1) mx = fmaxf(mx, __shfl_xor(mx, d, 32));
      const float mn = fmaxf(mrun[r], mx);
      const float a = __expf(mrun[r] - mn);
      const float p0 = __expf(x0 - mn);
      const float p1 = __expf(x1 - mn);
      float ps = p0 + p1;
#pragma unroll
      for (int d = 1; d < 16; d <<= 1) ps += __shfl_xor(ps, d, 32);
      lrun[r] = lrun[r] * a + ps;
      mrun[r] = mn;
      alpha[r] = a;
      P[(mrow + r) * 32 + m16] = (__bf16)p0;
      P[(mrow + r) * 32 + 16 + m16] = (__bf16)p1;
    }
#pragma unroll
    for (int j = 0; j < 8; ++j)
#pragma unroll
      for (int r = 0; r < 8; ++r) O[j][r] *= alpha[r];
    __syncthreads();
    v16bf pf = load_frag(&P[0], 32, lane);
#pragma unroll
    for (int j = 0; j < 8; ++j) {
      v16bf vf = load_frag(Vp + (size_t)(j * 16) * STOT + s0, STOT, lane);
      O[j] = wmma_bf16(pf, vf, O[j]);
    }
    __syncthreads();
  }

#pragma unroll
  for (int r = 0; r < 8; ++r) {
    const float inv = 1.0f / lrun[r];
#pragma unroll
    for (int j = 0; j < 8; ++j) {
      oa[(size_t)(b * TT + t0 + mrow + r) * (NQh * HH) + n * HH + hh * 128 +
         j * 16 + m16] = (__bf16)(O[j][r] * inv);
    }
  }
}

extern "C" void kernel_launch(void* const* d_in, const int* in_sizes, int n_in,
                              void* d_out, int out_size, void* d_ws, size_t ws_size,
                              hipStream_t stream) {
  (void)in_sizes; (void)n_in; (void)out_size; (void)ws_size;
  const float* hidden = (const float*)d_in[0];
  const float* enc    = (const float*)d_in[1];
  const int*   pos    = (const int*)d_in[2];
  // d_in[3] merged_attention_mask: computed analytically (causal + cross-visible)
  const float* q_w = (const float*)d_in[4];
  const float* k_w = (const float*)d_in[5];
  const float* v_w = (const float*)d_in[6];
  const float* o_w = (const float*)d_in[7];
  const float* q_sc = (const float*)d_in[8];
  const float* k_sc = (const float*)d_in[9];
  float* out = (float*)d_out;

  char* p = (char*)d_ws;
  auto alloc = [&](size_t bytes) -> void* {
    void* r = (void*)p;
    p += (bytes + 255) & ~(size_t)255;
    return r;
  };
  __bf16* hid_bf = (__bf16*)alloc((size_t)BB * TT * DD * 2);
  __bf16* enc_bf = (__bf16*)alloc((size_t)BB * SENC * DD * 2);
  __bf16* q_wt = (__bf16*)alloc((size_t)NQh * DD * HH * 2);
  __bf16* k_wt = (__bf16*)alloc((size_t)NKVh * DD * HH * 2);
  __bf16* v_wt = (__bf16*)alloc((size_t)NKVh * DD * HH * 2);
  __bf16* o_wt = (__bf16*)alloc((size_t)NQh * HH * DD * 2);
  float* q_f32 = (float*)alloc((size_t)BB * TT * NQh * HH * 4);
  float* sk_f32 = (float*)alloc((size_t)BB * TT * NKVh * HH * 4);
  float* sv_f32 = (float*)alloc((size_t)BB * TT * NKVh * HH * 4);
  float* ck_f32 = (float*)alloc((size_t)BB * SENC * NKVh * HH * 4);
  float* cv_f32 = (float*)alloc((size_t)BB * SENC * NKVh * HH * 4);
  __bf16* q_attn = (__bf16*)alloc((size_t)BB * NQh * TT * HH * 2);
  __bf16* k_attn = (__bf16*)alloc((size_t)BB * NKVh * STOT * HH * 2);
  __bf16* v_t = (__bf16*)alloc((size_t)BB * NKVh * HH * STOT * 2);
  __bf16* attn_o = (__bf16*)alloc((size_t)BB * TT * NQh * HH * 2);

  // 1) casts
  {
    int n = BB * TT * DD;
    cast_bf16_kernel<<<dim3((n + 255) / 256), dim3(256), 0, stream>>>(hidden, hid_bf, n);
    cast_bf16_kernel<<<dim3((n + 255) / 256), dim3(256), 0, stream>>>(enc, enc_bf, n);
  }
  // 2) weight transposes: [D,H] -> [H,D] per head; o_w [NQ*H, D] -> [D, NQ*H]
  transpose_cast_kernel<<<dim3((DD * HH + 255) / 256, 1, NQh), dim3(256), 0, stream>>>(
      q_w, q_wt, DD, HH);
  transpose_cast_kernel<<<dim3((DD * HH + 255) / 256, 1, NKVh), dim3(256), 0, stream>>>(
      k_w, k_wt, DD, HH);
  transpose_cast_kernel<<<dim3((DD * HH + 255) / 256, 1, NKVh), dim3(256), 0, stream>>>(
      v_w, v_wt, DD, HH);
  transpose_cast_kernel<<<dim3((NQh * HH * DD + 255) / 256, 1, 1), dim3(256), 0, stream>>>(
      o_w, o_wt, NQh * HH, DD);

  // 3) projection GEMMs: one wave = 64x32 tile; M = B*T = 2048 rows
  const int Mt = (BB * TT) / 64;  // 32
  const int Nt = HH / 32;         // 8
  gemm_bf16_wmma<<<dim3(Mt, Nt, NQh), dim3(32), 0, stream>>>(
      hid_bf, DD, 0, q_wt, DD, (size_t)HH * DD, q_f32, NQh * HH, (size_t)HH, DD);
  gemm_bf16_wmma<<<dim3(Mt, Nt, NKVh), dim3(32), 0, stream>>>(
      hid_bf, DD, 0, k_wt, DD, (size_t)HH * DD, sk_f32, NKVh * HH, (size_t)HH, DD);
  gemm_bf16_wmma<<<dim3(Mt, Nt, NKVh), dim3(32), 0, stream>>>(
      hid_bf, DD, 0, v_wt, DD, (size_t)HH * DD, sv_f32, NKVh * HH, (size_t)HH, DD);
  gemm_bf16_wmma<<<dim3(Mt, Nt, NKVh), dim3(32), 0, stream>>>(
      enc_bf, DD, 0, k_wt, DD, (size_t)HH * DD, ck_f32, NKVh * HH, (size_t)HH, DD);
  gemm_bf16_wmma<<<dim3(Mt, Nt, NKVh), dim3(32), 0, stream>>>(
      enc_bf, DD, 0, v_wt, DD, (size_t)HH * DD, cv_f32, NKVh * HH, (size_t)HH, DD);

  // 4) RMS-norm (+RoPE, +SCALE for q) epilogues -> bf16 attention operands
  qk_epilogue<<<dim3(TT, NQh, BB), dim3(HH), 0, stream>>>(
      q_f32, q_sc, pos, SCALEF, 1, q_attn, TT, 0);
  qk_epilogue<<<dim3(TT, NKVh, BB), dim3(HH), 0, stream>>>(
      sk_f32, k_sc, pos, 1.0f, 1, k_attn, STOT, 0);
  qk_epilogue<<<dim3(SENC, NKVh, BB), dim3(HH), 0, stream>>>(
      ck_f32, k_sc, pos, 1.0f, 0, k_attn, STOT, TT);

  // 5) V pack (concat + transpose to [b,kv,h,s])
  {
    size_t tot = (size_t)BB * STOT * NKVh * HH;
    v_pack_kernel<<<dim3((unsigned)((tot + 255) / 256)), dim3(256), 0, stream>>>(
        sv_f32, cv_f32, v_t);
  }

  // 6) flash attention (grid.z packs batch and h-half)
  attn_wmma<<<dim3(TT / 16, NQh, BB * 2), dim3(32), 0, stream>>>(
      q_attn, k_attn, v_t, attn_o);

  // 7) output projection: [B*T, NQ*H] x [NQ*H, D] -> f32 d_out
  gemm_bf16_wmma<<<dim3(Mt, DD / 32, 1), dim3(32), 0, stream>>>(
      attn_o, NQh * HH, 0, o_wt, NQh * HH, 0, out, DD, 0, NQh * HH);
}
